// BehaviorSpecificFeedForward_6073083757091
// MI455X (gfx1250) — compile-verified, matrix operand order checked by reading, not run on hardware
//
#include <hip/hip_runtime.h>
#include <math.h>

// ---------------------------------------------------------------------------
// BehaviorSpecificFeedForward (MoE-style, 4 experts + zero-padding expert)
//   out[tok] = LN(x + W2[e]·gelu(W1[e]·x + b1[e]) + b2[e]),  e = type-1
// Grouped-GEMM: bucket tokens by expert, 64-token tiles, bf16 WMMA
// (v_wmma_f32_16x16x32_bf16, f32 accumulate), FFN fused through LDS.
// Weight tiles are streamed by the Tensor Data Mover (tensor_load_to_lds,
// TENSORcnt) overlapping the opposite GEMM phase; LDS row padding is done
// by the TDM pad_interval/pad_amount feature.
// ---------------------------------------------------------------------------

#define H       512
#define FF      2048
#define NB      4
#define MT      64          // tokens per tile
#define FC      64          // F chunk per fused iteration
#define NCH     (FF / FC)   // 32 chunks

typedef __attribute__((ext_vector_type(8)))  float  v8f;
typedef __attribute__((ext_vector_type(8)))  __bf16 v8bf;
typedef __attribute__((ext_vector_type(16))) __bf16 v16bf;
typedef __attribute__((ext_vector_type(4)))  unsigned int v4u;
typedef __attribute__((ext_vector_type(8)))  int v8i;
typedef __attribute__((ext_vector_type(4)))  int v4i;

// LDS layout. Row strides padded +16B (done by TDM padding for W tiles).
#define XS_STRIDE   520     // elems: 64 rows x 512 bf16 (x tile)
#define WS1_STRIDE  520     // elems: 64 rows (fLocal) x 512 (k=h)  W1 chunk
#define WS2_STRIDE  72      // elems: 512 rows (n=h) x 64 (kLocal=f) W2 chunk
#define HS_STRIDE   72      // elems: 64 rows x 64 bf16 (gelu hidden chunk)
#define YS_STRIDE   516     // floats: 64 rows x 512 f32 pre-LN (overlays Ws1/Ws2)

#define IDX_OFF   0                               // 64 ints
#define XS_OFF    256
#define WS1_OFF   (XS_OFF + 64 * XS_STRIDE * 2)   // 66816
#define WS2_OFF   (WS1_OFF + 64 * WS1_STRIDE * 2) // 133376
#define HS_OFF    (WS2_OFF + 512 * WS2_STRIDE * 2)// 207104
#define SMEM_BYTES (HS_OFF + 64 * HS_STRIDE * 2)  // 216320 <= 320KB WGP LDS

// Workspace layout (bytes); total ~32.3 MB.
#define WOFF_COUNTS 0
#define WOFF_IDX    64
#define WOFF_XB(nt)   (64 + (size_t)NB*(nt)*4)
#define WOFF_W1T(nt)  ((WOFF_XB(nt) + (size_t)(nt)*H*2 + 255) & ~(size_t)255)
#define WOFF_W2T(nt)  (WOFF_W1T(nt) + (size_t)NB*H*FF*2)

__device__ __forceinline__ unsigned short f2bf(float f) {
  unsigned int u = __float_as_uint(f);
  unsigned int r = (u + 0x7FFFu + ((u >> 16) & 1u)) >> 16;   // round-nearest-even
  return (unsigned short)r;
}

__device__ __forceinline__ v16bf ldpair(const unsigned short* p) {
  // CDNA5 16-bit A/B operand: per lane, 8 bf16 at K-base and 8 bf16 at K-base+16.
  v8bf lo = *(const v8bf*)(p);
  v8bf hi = *(const v8bf*)(p + 16);
  return __builtin_shufflevector(lo, hi, 0,1,2,3,4,5,6,7,8,9,10,11,12,13,14,15);
}

// Tensor Data Mover: 2D bf16 tile global->LDS with per-row +16B LDS padding.
// D# encoding per CDNA5 ISA (groups 2/3 zero => 2D).
// 6-arg builtin form (clang-23 / therock-10.0): trailing int32x8 zero-filled.
__device__ __forceinline__ void tdm_load_2d(const void* gaddr, unsigned int lds_byte,
                                            unsigned int tile_d0, unsigned int tile_d1,
                                            unsigned long long stride0_elems,
                                            unsigned int pad_interval_code,
                                            unsigned int pad_amount_code) {
  unsigned long long ga = (unsigned long long)gaddr;
  v4u g0;
  g0[0] = 1u;                                               // count=1, user mode
  g0[1] = lds_byte;                                         // lds_addr (bytes)
  g0[2] = (unsigned int)(ga & 0xFFFFFFFFu);                 // global_addr[31:0]
  g0[3] = (unsigned int)((ga >> 32) & 0x1FFFFFFu) | (2u << 30); // addr[56:32] | type=2
  v8i g1;
  g1[0] = (int)((1u << 16) |                                // data_size = 2 bytes
                (1u << 20) |                                // pad_enable
                (pad_interval_code << 22) |
                (pad_amount_code << 25));
  g1[1] = (int)((tile_d0 & 0xFFFFu) << 16);                 // tensor_dim0 = tile_d0
  g1[2] = (int)((tile_d0 >> 16) | ((tile_d1 & 0xFFFFu) << 16)); // tensor_dim1 = tile_d1
  g1[3] = (int)((tile_d1 >> 16) | (tile_d0 << 16));         // tile_dim0
  g1[4] = (int)(tile_d1);                                   // tile_dim1 | tile_dim2=0
  g1[5] = (int)(stride0_elems & 0xFFFFFFFFull);             // tensor_dim0_stride lo
  g1[6] = (int)((unsigned int)(stride0_elems >> 32) & 0xFFFFu); // stride hi | stride1=0
  g1[7] = 0;
  v4i z4 = {0, 0, 0, 0};
  v8i z8 = {0, 0, 0, 0, 0, 0, 0, 0};
  __builtin_amdgcn_tensor_load_to_lds(g0, g1, z4, z4, z8, 0);
}

// ---------------- kernel 0: zero expert counters ---------------------------
__global__ void k_zero(int* counts) {
  if (threadIdx.x < 8) counts[threadIdx.x] = 0;
}

// ---------------- kernel 1: route tokens, x -> bf16, zero type-0 outputs ---
__global__ void k_gather(const float* __restrict__ x, const int* __restrict__ types,
                         unsigned short* __restrict__ Xb, int* __restrict__ counts,
                         int* __restrict__ idxlist, float* __restrict__ out, int ntok) {
  int tok  = blockIdx.x * 8 + (threadIdx.x >> 5);
  int lane = threadIdx.x & 31;
  if (tok >= ntok) return;
  int ty = types[tok];
  if (lane == 0 && ty > 0) {
    int p = atomicAdd(&counts[ty - 1], 1);
    idxlist[(size_t)(ty - 1) * ntok + p] = tok;
  }
  const float4* xr = (const float4*)(x + (size_t)tok * H);
#pragma unroll
  for (int i = 0; i < 4; ++i) {
    float4 v = xr[lane + i * 32];
    ushort4 b;
    b.x = f2bf(v.x); b.y = f2bf(v.y); b.z = f2bf(v.z); b.w = f2bf(v.w);
    *(ushort4*)(Xb + (size_t)tok * H + (size_t)(lane + i * 32) * 4) = b;
  }
  if (ty == 0) {
    float4 z = make_float4(0.f, 0.f, 0.f, 0.f);
    float4* orow = (float4*)(out + (size_t)tok * H);
#pragma unroll
    for (int i = 0; i < 4; ++i) orow[lane + i * 32] = z;
  }
}

// ---------------- kernel 2: weights f32 -> bf16, K-contiguous transposes ---
// W1 [e][h][f] -> W1t [e][f][h]   (GEMM1 B operand: fixed f, contiguous k=h)
// W2 [e][f][h] -> W2t [e][h][f]   (GEMM2 B operand: fixed h, contiguous k=f)
__global__ void k_wconv(const float* __restrict__ W1, const float* __restrict__ W2,
                        unsigned short* __restrict__ W1t, unsigned short* __restrict__ W2t) {
  size_t i = (size_t)blockIdx.x * 256 + threadIdx.x;       // 0 .. NB*H*FF-1
  size_t e = i / ((size_t)H * FF);
  size_t rem = i % ((size_t)H * FF);
  size_t h = rem / FF, f = rem % FF;                       // W1 linear = [e][h][f]
  W1t[e * (size_t)FF * H + f * H + h] = f2bf(W1[i]);
  size_t f2 = rem / H, h2 = rem % H;                       // W2 linear = [e][f][h]
  W2t[e * (size_t)H * FF + h2 * FF + f2] = f2bf(W2[i]);
}

// ---------------- kernel 3: fused grouped-GEMM FFN + LayerNorm -------------
__global__ __launch_bounds__(256)
void k_ffn(const float* __restrict__ x,
           const float* __restrict__ b1, const float* __restrict__ b2,
           const float* __restrict__ gamma, const float* __restrict__ beta,
           const unsigned short* __restrict__ Xb,
           const unsigned short* __restrict__ W1t, const unsigned short* __restrict__ W2t,
           const int* __restrict__ counts, const int* __restrict__ idxlist,
           float* __restrict__ out, int ntok) {
  extern __shared__ char smem[];
  int* tile_tok      = (int*)(smem + IDX_OFF);
  unsigned short* Xs = (unsigned short*)(smem + XS_OFF);
  unsigned short* W1s = (unsigned short*)(smem + WS1_OFF);
  unsigned short* W2s = (unsigned short*)(smem + WS2_OFF);
  unsigned short* Hs = (unsigned short*)(smem + HS_OFF);
  float* Ys          = (float*)(smem + WS1_OFF);   // overlays W1s+W2s after GEMMs

  const int e    = blockIdx.y;
  const int cnt  = counts[e];
  const int base = blockIdx.x * MT;
  if (base >= cnt) return;

  const int tid  = threadIdx.x;
  const int w    = tid >> 5;
  const int lane = tid & 31;
  const int ln15 = lane & 15;
  const int half8 = (lane >> 4) * 8;
  const int rb   = w & 3;                   // 16-row block owned by wave
  const size_t w1off = (size_t)e * FF * H;
  const size_t w2off = (size_t)e * H * FF;
  const unsigned lds0 = (unsigned)__builtin_amdgcn_groupstaticsize();

  // --- kick off W1 chunk 0 DMA immediately (overlaps X staging) ---
  if (w == 0) {
    // 64 rows x 512 elems, row 1024B=128 qwords -> pad code 7, +4 DWORDs -> code 3
    tdm_load_2d(W1t + w1off, lds0 + WS1_OFF, 512u, 64u, 512ull, 7u, 3u);
  }

  if (tid < MT) {
    int t = base + tid;
    tile_tok[tid] = (t < cnt) ? idxlist[(size_t)e * ntok + t] : -1;
  }
  __syncthreads();

  // prefetch f32 residual rows for the epilogue (global_prefetch_b8)
  if (tid < MT) {
    int tok = tile_tok[tid];
    if (tok >= 0) {
      __builtin_prefetch(x + (size_t)tok * H, 0, 1);
      __builtin_prefetch(x + (size_t)tok * H + 256, 0, 1);
    }
  }

  // --- stage X tile (bf16) into LDS: 64 rows x 512 ---
#pragma unroll
  for (int i = 0; i < 16; ++i) {
    int lin = tid + i * 256;
    int r = lin >> 6, c = lin & 63;
    int tok = tile_tok[r];
    uint4 v = make_uint4(0u, 0u, 0u, 0u);
    if (tok >= 0) v = *(const uint4*)(Xb + (size_t)tok * H + (size_t)c * 8);
    *(uint4*)(Xs + (size_t)r * XS_STRIDE + c * 8) = v;
  }

  const v8f vzero = {};
  v8f yacc[16];
#pragma unroll
  for (int j = 0; j < 16; ++j) yacc[j] = vzero;

  if (w == 0) __builtin_amdgcn_s_wait_tensorcnt(0);   // W1(0) in LDS
  __syncthreads();

  // ---------------- fused FFN chunk loop over F ----------------
  for (int ch = 0; ch < NCH; ++ch) {
    const int kf = ch * FC;

    // DMA W2(ch) while GEMM1(ch) computes (W2s free: barrier at loop end)
    if (w == 0) {
      // 512 rows x 64 elems, row 128B=16 qwords -> pad code 4, +4 DWORDs -> code 3
      tdm_load_2d(W2t + w2off + kf, lds0 + WS2_OFF, 64u, 512u, 2048ull, 4u, 3u);
    }

    // GEMM1: h[64 x 64] = Xs[64 x 512] * W1chunk ; wave: 16 rows x 32 cols
    const int cb0 = (w >> 2) * 32;
    v8f hacc[2];
    hacc[0] = vzero; hacc[1] = vzero;
#pragma unroll
    for (int kk = 0; kk < 16; ++kk) {
      const int k0 = kk * 32;
      v16bf a  = ldpair(Xs + (size_t)(rb * 16 + ln15) * XS_STRIDE + k0 + half8);
      v16bf b0 = ldpair(W1s + (size_t)(cb0 + ln15) * WS1_STRIDE + k0 + half8);
      v16bf b1 = ldpair(W1s + (size_t)(cb0 + 16 + ln15) * WS1_STRIDE + k0 + half8);
      hacc[0] = __builtin_amdgcn_wmma_f32_16x16x32_bf16(
          false, a, false, b0, (short)0, hacc[0], false, false);
      hacc[1] = __builtin_amdgcn_wmma_f32_16x16x32_bf16(
          false, a, false, b1, (short)0, hacc[1], false, false);
    }

    // +b1, exact GELU, -> bf16 hidden chunk in LDS
#pragma unroll
    for (int j = 0; j < 2; ++j) {
      const int cloc = cb0 + j * 16 + ln15;
      const float bias = b1[(size_t)e * FF + kf + cloc];
#pragma unroll
      for (int jj = 0; jj < 8; ++jj) {
        const int m = rb * 16 + jj + half8;
        float v = hacc[j][jj] + bias;
        float g = 0.5f * v * (1.0f + erff(v * 0.70710678118654752f));
        Hs[(size_t)m * HS_STRIDE + cloc] = f2bf(g);
      }
    }

    if (w == 0) __builtin_amdgcn_s_wait_tensorcnt(0);   // W2(ch) in LDS
    __syncthreads();                                    // Hs + W2s visible, W1s free

    // DMA W1(ch+1) while GEMM2(ch) computes
    if (w == 0 && ch + 1 < NCH) {
      tdm_load_2d(W1t + w1off + (size_t)(kf + FC) * H, lds0 + WS1_OFF,
                  512u, 64u, 512ull, 7u, 3u);
    }

    // GEMM2 accumulate: y[64 x 512] += Hs[64 x 64] * W2chunk ; wave: 16 x 256
    const int nc0 = (w >> 2) * 256;
#pragma unroll
    for (int kk = 0; kk < 2; ++kk) {
      const int k0 = kk * 32;
      v16bf a = ldpair(Hs + (size_t)(rb * 16 + ln15) * HS_STRIDE + k0 + half8);
      v16bf b = ldpair(W2s + (size_t)(nc0 + ln15) * WS2_STRIDE + k0 + half8);
#pragma unroll
      for (int j = 0; j < 16; ++j) {
        v16bf bn = b;   // rotate: next B load in flight behind current WMMA
        if (j < 15)
          bn = ldpair(W2s + (size_t)(nc0 + (j + 1) * 16 + ln15) * WS2_STRIDE + k0 + half8);
        yacc[j] = __builtin_amdgcn_wmma_f32_16x16x32_bf16(
            false, a, false, b, (short)0, yacc[j], false, false);
        b = bn;
      }
    }

    if (w == 0) __builtin_amdgcn_s_wait_tensorcnt(0);   // W1(ch+1) in LDS
    __syncthreads();                                    // W2s/Hs free for next chunk
  }

  // epilogue: y + b2 + x(residual, f32) -> Ys (overlays W1s/W2s)
  {
    const int nc0 = (w >> 2) * 256;
#pragma unroll
    for (int j = 0; j < 16; ++j) {
      const int ncol = nc0 + j * 16 + ln15;
      const float bv = b2[(size_t)e * H + ncol];
#pragma unroll
      for (int jj = 0; jj < 8; ++jj) {
        const int r = rb * 16 + jj + half8;
        const int tok = tile_tok[r];
        float xres = (tok >= 0) ? x[(size_t)tok * H + ncol] : 0.f;
        Ys[(size_t)r * YS_STRIDE + ncol] = yacc[j][jj] + bv + xres;
      }
    }
  }
  __syncthreads();

  // LayerNorm per row (8 rows per wave), write gathered output
  for (int i = 0; i < 8; ++i) {
    const int r = w * 8 + i;
    const int tok = tile_tok[r];
    if (tok < 0) continue;
    float v[16];
    float s = 0.f;
#pragma unroll
    for (int k = 0; k < 16; ++k) {
      v[k] = Ys[(size_t)r * YS_STRIDE + k * 32 + lane];
      s += v[k];
    }
#pragma unroll
    for (int m = 16; m >= 1; m >>= 1) s += __shfl_xor(s, m, 32);
    const float mean = s * (1.0f / (float)H);
    float q = 0.f;
#pragma unroll
    for (int k = 0; k < 16; ++k) { float d = v[k] - mean; q += d * d; }
#pragma unroll
    for (int m = 16; m >= 1; m >>= 1) q += __shfl_xor(q, m, 32);
    const float rstd = rsqrtf(q * (1.0f / (float)H) + 1e-12f);
#pragma unroll
    for (int k = 0; k < 16; ++k) {
      const int col = k * 32 + lane;
      out[(size_t)tok * H + col] =
          (v[k] - mean) * rstd * gamma[(size_t)e * H + col] + beta[(size_t)e * H + col];
    }
  }
}

// ---------------------------------------------------------------------------
extern "C" void kernel_launch(void* const* d_in, const int* in_sizes, int n_in,
                              void* d_out, int out_size, void* d_ws, size_t ws_size,
                              hipStream_t stream) {
  const float* x     = (const float*)d_in[0];
  const int*   types = (const int*)d_in[1];
  const float* W1    = (const float*)d_in[2];
  const float* b1    = (const float*)d_in[3];
  const float* W2    = (const float*)d_in[4];
  const float* b2    = (const float*)d_in[5];
  const float* gamma = (const float*)d_in[6];
  const float* beta  = (const float*)d_in[7];
  float* out = (float*)d_out;

  const int ntok = in_sizes[0] / H;          // 16384 for the reference shapes

  char* ws = (char*)d_ws;
  int* counts         = (int*)(ws + WOFF_COUNTS);
  int* idxlist        = (int*)(ws + WOFF_IDX);
  unsigned short* Xb  = (unsigned short*)(ws + WOFF_XB(ntok));
  unsigned short* W1t = (unsigned short*)(ws + WOFF_W1T(ntok));
  unsigned short* W2t = (unsigned short*)(ws + WOFF_W2T(ntok));
  // requires ws_size >= WOFF_W2T(ntok) + NB*H*FF*2  (~32.3 MB at ntok=16384)

  k_zero<<<1, 32, 0, stream>>>(counts);
  k_gather<<<(ntok + 7) / 8, 256, 0, stream>>>(x, types, Xb, counts, idxlist, out, ntok);
  k_wconv<<<(NB * H * FF) / 256, 256, 0, stream>>>(W1, W2, W1t, W2t);

  const int tiles = (ntok + MT - 1) / MT;
  k_ffn<<<dim3(tiles, NB), 256, SMEM_BYTES, stream>>>(
      x, b1, b2, gamma, beta, Xb, W1t, W2t, counts, idxlist, out, ntok);
}
